// NonLocalAttention_36455682409121
// MI455X (gfx1250) — matrix operand order repaired and checked by reading, not compiled, not run on hardware
//
#include <hip/hip_runtime.h>

typedef __attribute__((ext_vector_type(16))) _Float16 v16h;
typedef __attribute__((ext_vector_type(8)))  _Float16 v8h;
typedef __attribute__((ext_vector_type(8)))  float    v8f;

#define NB 2
#define CC 64
#define CR 32
#define HWP 9216   // 96*96

// ---------------------------------------------------------------------------
// Kernel 1: 1x1 convs + PReLU.  Produces:
//   Qh [N][HW][32]  f16 (query-major rows of 32 channels)
//   Kh [N][HW][32]  f16 (key-major rows of 32 channels)
//   Vt [N][64][HW]  f16 (channel-major, key-contiguous)
// ---------------------------------------------------------------------------
__global__ __launch_bounds__(256) void nla_prep_kernel(
    const float* __restrict__ inputa, const float* __restrict__ inputb,
    const float* __restrict__ w1, const float* __restrict__ b1, const float* __restrict__ a1,
    const float* __restrict__ w2, const float* __restrict__ b2, const float* __restrict__ a2,
    const float* __restrict__ wa, const float* __restrict__ ba, const float* __restrict__ aa,
    _Float16* __restrict__ Qh, _Float16* __restrict__ Kh, _Float16* __restrict__ Vt)
{
    __shared__ float lw1[CR * CC];
    __shared__ float lw2[CR * CC];
    __shared__ float lwa[CC * CC];
    __shared__ float lb1[CR], lb2[CR], lba[CC];

    const int tid = threadIdx.x;
    for (int i = tid; i < CR * CC; i += 256) { lw1[i] = w1[i]; lw2[i] = w2[i]; }
    for (int i = tid; i < CC * CC; i += 256) { lwa[i] = wa[i]; }
    if (tid < CR) { lb1[tid] = b1[tid]; lb2[tid] = b2[tid]; }
    if (tid < CC) { lba[tid] = ba[tid]; }
    __syncthreads();

    const float s1 = a1[0], s2 = a2[0], sa = aa[0];

    const int p   = blockIdx.x * 256 + tid;   // 0 .. N*HW-1 (exact)
    const int n   = p / HWP;
    const int pix = p - n * HWP;

    float x[CC];
    const float* xa = inputa + (size_t)n * CC * HWP + pix;
    #pragma unroll
    for (int c = 0; c < CC; ++c) x[c] = xa[(size_t)c * HWP];

    _Float16* qdst = Qh + ((size_t)n * HWP + pix) * CR;
    #pragma unroll 4
    for (int o = 0; o < CR; ++o) {
        float acc = lb1[o];
        #pragma unroll
        for (int c = 0; c < CC; ++c) acc = __fmaf_rn(lw1[o * CC + c], x[c], acc);
        acc = acc >= 0.0f ? acc : s1 * acc;
        qdst[o] = (_Float16)acc;
    }
    #pragma unroll 4
    for (int o = 0; o < CC; ++o) {
        float acc = lba[o];
        #pragma unroll
        for (int c = 0; c < CC; ++c) acc = __fmaf_rn(lwa[o * CC + c], x[c], acc);
        acc = acc >= 0.0f ? acc : sa * acc;
        Vt[((size_t)n * CC + o) * HWP + pix] = (_Float16)acc;
    }
    const float* xb = inputb + (size_t)n * CC * HWP + pix;
    #pragma unroll
    for (int c = 0; c < CC; ++c) x[c] = xb[(size_t)c * HWP];
    _Float16* kdst = Kh + ((size_t)n * HWP + pix) * CR;
    #pragma unroll 4
    for (int o = 0; o < CR; ++o) {
        float acc = lb2[o];
        #pragma unroll
        for (int c = 0; c < CC; ++c) acc = __fmaf_rn(lw2[o * CC + c], x[c], acc);
        acc = acc >= 0.0f ? acc : s2 * acc;
        kdst[o] = (_Float16)acc;
    }
}

// ---------------------------------------------------------------------------
// 16-lane xor-butterfly max.  Prefer v_permlane16_b32 (pure VALU) over
// ds_bpermute (LDS pipe).  Nibble selects implement lane ^= {1,2,4,8}
// within each 16-lane group.
// ---------------------------------------------------------------------------
#if __has_builtin(__builtin_amdgcn_permlane16)
__device__ __forceinline__ float pl16(float v, unsigned lo, unsigned hi) {
    unsigned s = __float_as_uint(v);
    unsigned r = __builtin_amdgcn_permlane16(s, s, lo, hi, false, false);
    return __uint_as_float(r);
}
__device__ __forceinline__ float rmax16(float v) {
    v = fmaxf(v, pl16(v, 0x67452301u, 0xEFCDAB89u));  // xor 1
    v = fmaxf(v, pl16(v, 0x54761032u, 0xDCFE98BAu));  // xor 2
    v = fmaxf(v, pl16(v, 0x32107654u, 0xBA98FEDCu));  // xor 4
    v = fmaxf(v, pl16(v, 0xFEDCBA98u, 0x76543210u));  // xor 8
    return v;
}
#else
__device__ __forceinline__ float rmax16(float v) {
    v = fmaxf(v, __shfl_xor(v, 1, 32));
    v = fmaxf(v, __shfl_xor(v, 2, 32));
    v = fmaxf(v, __shfl_xor(v, 4, 32));
    v = fmaxf(v, __shfl_xor(v, 8, 32));
    return v;
}
#endif

// ---------------------------------------------------------------------------
// Kernel 2: fused flash attention + residual, BN = 128 keys per step.
// Per step: 8 WMMAs for S, 1 register-space tile max + ONE lane butterfly
// per row, exp, P->LDS transpose, 4 WMMAs (P x ones) for the row sums,
// 16 WMMAs for P*V (C = 64).
// ---------------------------------------------------------------------------
__global__ __launch_bounds__(256) void nla_attn_kernel(
    const _Float16* __restrict__ Qh, const _Float16* __restrict__ Kh,
    const _Float16* __restrict__ Vt, const float* __restrict__ inputa,
    float* __restrict__ out)
{
    __shared__ __align__(64) _Float16 Pl[8][16][128];   // 32 KB: per-wave P staging

    const int lane = threadIdx.x & 31;
    const int wv   = threadIdx.x >> 5;
    const int tile = blockIdx.x * 8 + wv;              // 1152 tiles, 144 blocks exact
    const int n     = tile / (HWP / 16);
    const int qbase = (tile - n * (HWP / 16)) * 16;

    const int col = lane & 15;     // N-column (B/C layouts) / row for A-frag gathers
    const int hh  = lane >> 4;     // half-wave select

    union V16 { v16h v; v8h h[2]; };

    // ---- Q A-fragment
    V16 aq;
    const _Float16* qrow = Qh + ((size_t)n * HWP + qbase + col) * CR;
    aq.v = *(const v16h*)(qrow + 16 * hh);

    // ---- ones B-fragment (32x16 of 1.0) for row-sum WMMAs
    v16h onesv;
    #pragma unroll
    for (int i = 0; i < 16; ++i) onesv[i] = (_Float16)1.0f;

    float m[8], l[8];
    v8f acc0 = {}, acc1 = {}, acc2 = {}, acc3 = {};
    #pragma unroll
    for (int j = 0; j < 8; ++j) { m[j] = -__builtin_inff(); l[j] = 0.0f; }

    const _Float16* kb_ptr = Kh + (size_t)n * HWP * CR;
    const _Float16* vb_ptr = Vt + (size_t)n * CC * HWP;

    for (int kb = 0; kb < HWP; kb += 128) {
        // ---- 8 S-tiles: S[t] = Q x K[kb+16t .. kb+16t+15]
        v8f s[8];
        #pragma unroll
        for (int t = 0; t < 8; ++t) {
            v16h bk = *(const v16h*)(kb_ptr + (size_t)(kb + 16 * t + col) * CR + 16 * hh);
            v8f z = {};
            s[t] = __builtin_amdgcn_wmma_f32_16x16x32_f16(false, aq.v, false, bk,
                                                          (short)0, z, false, false);
        }
        // prefetch next key/value block (L2-resident; hides latency)
        if (kb + 128 < HWP) {
            __builtin_prefetch(kb_ptr + (size_t)(kb + 128 + col) * CR, 0, 3);
            __builtin_prefetch(vb_ptr + (size_t)col * HWP + kb + 128, 0, 3);
        }

        // ---- online softmax: one butterfly per row per 128 keys
        #pragma unroll
        for (int j = 0; j < 8; ++j) {
            float rm = s[0][j];
            #pragma unroll
            for (int t = 1; t < 8; ++t) rm = fmaxf(rm, s[t][j]);
            rm = rmax16(rm);
            const float mn = fmaxf(m[j], rm);
            const float sc = __expf(m[j] - mn);
            m[j] = mn;
            l[j] *= sc;
            acc0[j] *= sc; acc1[j] *= sc; acc2[j] *= sc; acc3[j] *= sc;
            #pragma unroll
            for (int t = 0; t < 8; ++t) {
                Pl[wv][j + 8 * hh][16 * t + col] = (_Float16)__expf(s[t][j] - mn);
            }
        }
        __asm__ volatile("s_wait_dscnt 0" ::: "memory");  // wave-local LDS transpose

        // ---- P A-fragments (4 x 32 keys) + row sums via ones-WMMA
        V16 ap[4];
        #pragma unroll
        for (int u = 0; u < 4; ++u) {
            ap[u].v = *(const v16h*)(&Pl[wv][col][32 * u + 16 * hh]);
        }
        v8f lsum = {};
        #pragma unroll
        for (int u = 0; u < 4; ++u) {
            lsum = __builtin_amdgcn_wmma_f32_16x16x32_f16(false, ap[u].v, false, onesv,
                                                          (short)0, lsum, false, false);
        }
        #pragma unroll
        for (int j = 0; j < 8; ++j) l[j] += lsum[j];

        // ---- P x V accumulation (4 key sub-tiles x 4 channel groups)
        #pragma unroll
        for (int u = 0; u < 4; ++u) {
            const int ko = kb + 32 * u + 16 * hh;
            v16h bv0 = *(const v16h*)(vb_ptr + (size_t)(col)      * HWP + ko);
            v16h bv1 = *(const v16h*)(vb_ptr + (size_t)(16 + col) * HWP + ko);
            v16h bv2 = *(const v16h*)(vb_ptr + (size_t)(32 + col) * HWP + ko);
            v16h bv3 = *(const v16h*)(vb_ptr + (size_t)(48 + col) * HWP + ko);
            acc0 = __builtin_amdgcn_wmma_f32_16x16x32_f16(false, ap[u].v, false, bv0, (short)0, acc0, false, false);
            acc1 = __builtin_amdgcn_wmma_f32_16x16x32_f16(false, ap[u].v, false, bv1, (short)0, acc1, false, false);
            acc2 = __builtin_amdgcn_wmma_f32_16x16x32_f16(false, ap[u].v, false, bv2, (short)0, acc2, false, false);
            acc3 = __builtin_amdgcn_wmma_f32_16x16x32_f16(false, ap[u].v, false, bv3, (short)0, acc3, false, false);
        }
    }

    // ---- epilogue: out[n][ch][q] = acc/l + inputa[n][ch][q]
    const float* ina  = inputa + (size_t)n * CC * HWP;
    float*       outp = out    + (size_t)n * CC * HWP;
    #pragma unroll
    for (int j = 0; j < 8; ++j) {
        const int q   = qbase + j + 8 * hh;
        const float r = 1.0f / l[j];
        outp[(size_t)(col)      * HWP + q] = acc0[j] * r + ina[(size_t)(col)      * HWP + q];
        outp[(size_t)(16 + col) * HWP + q] = acc1[j] * r + ina[(size_t)(16 + col) * HWP + q];
        outp[(size_t)(32 + col) * HWP + q] = acc2[j] * r + ina[(size_t)(32 + col) * HWP + q];
        outp[(size_t)(48 + col) * HWP + q] = acc3[j] * r + ina[(size_t)(48 + col) * HWP + q];
    }
}

// ---------------------------------------------------------------------------
extern "C" void kernel_launch(void* const* d_in, const int* in_sizes, int n_in,
                              void* d_out, int out_size, void* d_ws, size_t ws_size,
                              hipStream_t stream) {
    const float* inputa = (const float*)d_in[0];
    const float* inputb = (const float*)d_in[1];
    const float* w1 = (const float*)d_in[2];
    const float* b1 = (const float*)d_in[3];
    const float* a1 = (const float*)d_in[4];
    const float* w2 = (const float*)d_in[5];
    const float* b2 = (const float*)d_in[6];
    const float* a2 = (const float*)d_in[7];
    const float* wa = (const float*)d_in[8];
    const float* ba = (const float*)d_in[9];
    const float* aa = (const float*)d_in[10];
    float* out = (float*)d_out;

    const size_t qk_bytes = (size_t)NB * HWP * CR * sizeof(_Float16); // 1,179,648
    char* ws = (char*)d_ws;
    _Float16* Qh = (_Float16*)(ws);
    _Float16* Kh = (_Float16*)(ws + qk_bytes);
    _Float16* Vt = (_Float16*)(ws + 2 * qk_bytes);                    // 2,359,296 B

    nla_prep_kernel<<<(NB * HWP) / 256, 256, 0, stream>>>(
        inputa, inputb, w1, b1, a1, w2, b2, a2, wa, ba, aa, Qh, Kh, Vt);

    nla_attn_kernel<<<(NB * (HWP / 16)) / 8, 256, 0, stream>>>(
        Qh, Kh, Vt, inputa, out);
}
